// GatedDeltaNet_20976620274204
// MI455X (gfx1250) — matrix operand order repaired and checked by reading, not compiled, not run on hardware
//
#include <hip/hip_runtime.h>

// ---------------- CDNA5 WMMA types ----------------
typedef __bf16 bf16;
typedef __attribute__((ext_vector_type(16))) __bf16 v16bf;
typedef __attribute__((ext_vector_type(8)))  __bf16 v8bf;
typedef __attribute__((ext_vector_type(8)))  float  v8f;

#define CHUNK 64      // chunk length (time)
#define DK    128     // key dim
#define DV    128     // full value dim
#define DVT   64      // value dim handled per workgroup (Dv split by 2)
#define NTHREADS 256  // 8 wave32

__device__ __forceinline__ v8f zero8() {
    v8f z = {0.f,0.f,0.f,0.f,0.f,0.f,0.f,0.f};
    return z;
}

__device__ __forceinline__ v8f wmma_bf16(v16bf a, v16bf b, v8f c) {
    // D = A(16x32 bf16) * B(32x16 bf16) + C(16x16 f32)
    return __builtin_amdgcn_wmma_f32_16x16x32_bf16(
        /*neg_a=*/false, a, /*neg_b=*/false, b,
        /*c_mod=*/(short)0, c, /*reuse_a=*/false, /*reuse_b=*/false);
}

__device__ __forceinline__ v16bf cat8(v8bf lo, v8bf hb) {
    v16bf r;
#pragma unroll
    for (int i = 0; i < 8; ++i) { r[i] = lo[i]; r[8 + i] = hb[i]; }
    return r;
}

// A operand (16x32 bf16), src row-major [rows x ld], contiguous in K.
// lanes 0-15: row M=lane, V0-3: K=0..7, V4-7: K=16..23;
// lanes 16-31: row M=lane-16, V0-3: K=8..15, V4-7: K=24..31.
__device__ __forceinline__ v16bf load_a_rows(const bf16* src, int ld,
                                             int row0, int kk, int lane) {
    int hi = lane >> 4;
    int m  = lane & 15;
    const bf16* p = src + (row0 + m) * ld + kk + hi * 8;
    v8bf lo = *(const v8bf*)p;          // 16B aligned
    v8bf hb = *(const v8bf*)(p + 16);
    return cat8(lo, hb);
}

// B operand (32x16, K x N) where src stores B^T row-major: [N x ld], contiguous in K.
// lanes 0-15: N=lane, V0-3: K=0..7, V4-7: K=16..23; lanes 16-31: N=lane-16, V0-3: K=8..15, V4-7: K=24..31.
__device__ __forceinline__ v16bf load_b_rows(const bf16* src, int ld,
                                             int col0, int kk, int lane) {
    int hi = lane >> 4;
    int n  = lane & 15;
    const bf16* p = src + (col0 + n) * ld + kk + hi * 8;
    v8bf lo = *(const v8bf*)p;
    v8bf hb = *(const v8bf*)(p + 16);
    return cat8(lo, hb);
}

__global__ __launch_bounds__(NTHREADS)
void gdn_chunked_wmma_kernel(const float* __restrict__ gq,
                             const float* __restrict__ gk,
                             const float* __restrict__ gv,
                             const float* __restrict__ galpha,
                             const float* __restrict__ gbeta,
                             const float* __restrict__ gs0,
                             float* __restrict__ gout,
                             float* __restrict__ gsfinal) {
    constexpr int B = 2, T = 4096, H = 16;

    // ---- LDS (~146 KB, fits 320 KB WGP budget) ----
    __shared__ float S   [DK * DVT];                  // fp32 running state (Dk x DvT)
    __shared__ __align__(16) bf16 SbT [DVT * DK];     // state^T bf16 (B operand)
    __shared__ __align__(16) bf16 Kb  [CHUNK * DK];   // chunk K, bf16 (A operand)
    __shared__ __align__(16) bf16 Qb  [CHUNK * DK];   // chunk Q, bf16 (A operand)
    __shared__ __align__(16) bf16 Ktld[DK * CHUNK];   // (K-tilde)^T pre-scaled (A operand)
    __shared__ float U   [CHUNK * DVT];               // rhs -> solved U (fp32)
    __shared__ __align__(16) bf16 UbT [DVT * CHUNK];  // U^T bf16 (B operand)
    __shared__ float Amat[CHUNK * CHUNK];             // strictly-lower A
    __shared__ __align__(16) bf16 Pb  [CHUNK * CHUNK];// masked, decayed Q@K^T (A operand)
    __shared__ float Lc[CHUNK], Gc[CHUNK], Gi[CHUNK], Bc[CHUNK], Rl[CHUNK];

    const int blk   = blockIdx.x;
    const int split = blk & 1;           // which half of Dv
    const int bh    = blk >> 1;
    const int b     = bh / H;
    const int h     = bh % H;
    const int dv0   = split * DVT;
    const int tid   = threadIdx.x;
    const int lane  = tid & 31;
    const int wave  = tid >> 5;
    const int n15   = lane & 15;
    const int hi    = lane >> 4;

    // ---- load initial state (vectorized) ----
    {
        const float* s0p = gs0 + ((size_t)(b * H + h)) * DK * DV + dv0;
        for (int i = tid; i < DK * DVT / 4; i += NTHREADS) {
            int d  = i / (DVT / 4);
            int vq = (i % (DVT / 4)) * 4;
            float4 x = *(const float4*)(s0p + (size_t)d * DV + vq);
            float* sp = S + d * DVT + vq;
            sp[0] = x.x; sp[1] = x.y; sp[2] = x.z; sp[3] = x.w;
            SbT[(vq + 0) * DK + d] = (bf16)x.x;
            SbT[(vq + 1) * DK + d] = (bf16)x.y;
            SbT[(vq + 2) * DK + d] = (bf16)x.z;
            SbT[(vq + 3) * DK + d] = (bf16)x.w;
        }
    }
    __syncthreads();

    for (int c0 = 0; c0 < T; c0 += CHUNK) {
        // ---- phase 1: gates, cumulative log-decay ----
        if (tid < CHUNK) {
            size_t gi = ((size_t)b * T + (c0 + tid)) * H + h;
            Lc[tid] = __logf(galpha[gi]);
            Bc[tid] = gbeta[gi];
        }
        __syncthreads();
        if (tid == 0) {
            float acc = 0.f;
            for (int t = 0; t < CHUNK; ++t) { acc += Lc[t]; Lc[t] = acc; }
        }
        __syncthreads();
        if (tid < CHUNK) {
            float l = Lc[tid];
            Gc[tid] = __expf(l);                    // g_t
            Gi[tid] = __expf(-l);                   // 1/g_t   (|l| <= ~45, safe in f32)
            Rl[tid] = __expf(Lc[CHUNK - 1] - l);    // g_last/g_t
        }
        __syncthreads();   // Rl needed by Ktld staging below

        // ---- phase 2: stage K, Q (bf16) and pre-scaled K~^T (vectorized) ----
        for (int i = tid; i < CHUNK * DK / 4; i += NTHREADS) {
            int t = i / (DK / 4);
            int d = (i % (DK / 4)) * 4;
            size_t gi = (((size_t)b * T + (c0 + t)) * H + h) * DK + d;
            float4 kv = *(const float4*)(gk + gi);
            float4 qv = *(const float4*)(gq + gi);
            bf16* kp = Kb + t * DK + d;
            kp[0] = (bf16)kv.x; kp[1] = (bf16)kv.y;
            kp[2] = (bf16)kv.z; kp[3] = (bf16)kv.w;
            bf16* qp = Qb + t * DK + d;
            qp[0] = (bf16)qv.x; qp[1] = (bf16)qv.y;
            qp[2] = (bf16)qv.z; qp[3] = (bf16)qv.w;
            float rl = Rl[t];
            Ktld[(d + 0) * CHUNK + t] = (bf16)(kv.x * rl);
            Ktld[(d + 1) * CHUNK + t] = (bf16)(kv.y * rl);
            Ktld[(d + 2) * CHUNK + t] = (bf16)(kv.z * rl);
            Ktld[(d + 3) * CHUNK + t] = (bf16)(kv.w * rl);
        }
        __syncthreads();

        // ---- phase 3: K@K^T -> Amat, Q@K^T -> Pb (masked + decayed) ----
#pragma unroll
        for (int ti = 0; ti < 2; ++ti) {
            int tile = wave * 2 + ti;
            int tm = (tile >> 2) * 16, tn = (tile & 3) * 16;
            v8f accK = zero8(), accQ = zero8();
#pragma unroll
            for (int kk = 0; kk < DK; kk += 32) {
                v16bf bK = load_b_rows(Kb, DK, tn, kk, lane);   // K^T as B
                v16bf aK = load_a_rows(Kb, DK, tm, kk, lane);
                v16bf aQ = load_a_rows(Qb, DK, tm, kk, lane);
                accK = wmma_bf16(aK, bK, accK);
                accQ = wmma_bf16(aQ, bK, accQ);
            }
            int s = tn + n15;
            float gis = Gi[s];
#pragma unroll
            for (int j = 0; j < 8; ++j) {
                int t = tm + j + hi * 8;
                float dec = Gc[t] * gis;            // exp(Lc[t]-Lc[s]), t>=s => <=1
                Amat[t * CHUNK + s] = (s < t) ? Bc[t] * dec * accK[j] : 0.f;
                Pb  [t * CHUNK + s] = (bf16)((s <= t) ? dec * accQ[j] : 0.f);
            }
        }
        __syncthreads();

        // ---- phase 4: Bpre = K @ S0 ; U = beta*(v - g*Bpre) ----
#pragma unroll
        for (int ti = 0; ti < 2; ++ti) {
            int tile = wave * 2 + ti;
            int tm = (tile >> 2) * 16, tn = (tile & 3) * 16;
            v8f acc = zero8();
#pragma unroll
            for (int kk = 0; kk < DK; kk += 32) {
                v16bf a  = load_a_rows(Kb, DK, tm, kk, lane);
                v16bf bb = load_b_rows(SbT, DK, tn, kk, lane);
                acc = wmma_bf16(a, bb, acc);
            }
#pragma unroll
            for (int j = 0; j < 8; ++j) {
                int t = tm + j + hi * 8, vc = tn + n15;
                size_t gi = (((size_t)b * T + (c0 + t)) * H + h) * DV + dv0 + vc;
                U[t * DVT + vc] = Bc[t] * (gv[gi] - Gc[t] * acc[j]);
            }
        }
        __syncthreads();

        // ---- phase 5: blocked forward substitution (I + A) U = B ----
        // 16-row blocks: register-resident diagonal solves + WMMA off-diag updates.
#pragma unroll
        for (int ib = 0; ib < 4; ++ib) {
            if (tid < DVT) {
                float u[16];
#pragma unroll
                for (int t = 0; t < 16; ++t) u[t] = U[(ib * 16 + t) * DVT + tid];
#pragma unroll
                for (int s = 0; s < 15; ++s) {
#pragma unroll
                    for (int t = s + 1; t < 16; ++t)
                        u[t] -= Amat[(ib * 16 + t) * CHUNK + ib * 16 + s] * u[s];
                }
#pragma unroll
                for (int t = 0; t < 16; ++t)
                    UbT[tid * CHUNK + ib * 16 + t] = (bf16)u[t];
            }
            __syncthreads();
            if (ib == 3) break;
            // U[jb] -= A[jb][ib] @ U[ib] for jb > ib  (K padded 16->32 with zeros)
            int ntiles = (3 - ib) * 4;
            if (wave < ntiles) {
                int jb = ib + 1 + (wave >> 2);
                int tm = jb * 16;
                int tn = (wave & 3) * 16;
                v8f c;
#pragma unroll
                for (int j = 0; j < 8; ++j)
                    c[j] = U[(tm + j + hi * 8) * DVT + tn + n15];
                v16bf a;   // -A block, rows tm.., K = cols of block ib (16 valid)
#pragma unroll
                for (int i2 = 0; i2 < 8; ++i2) {
                    a[i2]     = (bf16)(-Amat[(tm + n15) * CHUNK + ib * 16 + hi * 8 + i2]);
                    a[8 + i2] = (bf16)0.f;
                }
                v16bf bb;  // U[ib] in bf16 from UbT; elements 8-15 (K=16..31) are zero
#pragma unroll
                for (int i2 = 0; i2 < 8; ++i2) {
                    bb[i2]     = UbT[(tn + n15) * CHUNK + ib * 16 + hi * 8 + i2];
                    bb[8 + i2] = (bf16)0.f;
                }
                c = wmma_bf16(a, bb, c);
#pragma unroll
                for (int j = 0; j < 8; ++j)
                    U[(tm + j + hi * 8) * DVT + tn + n15] = c[j];
            }
            __syncthreads();
        }

        // ---- phase 7: O = g_t*(Q@S0) + P@U ----
#pragma unroll
        for (int ti = 0; ti < 2; ++ti) {
            int tile = wave * 2 + ti;
            int tm = (tile >> 2) * 16, tn = (tile & 3) * 16;
            v8f acc = zero8();
#pragma unroll
            for (int kk = 0; kk < DK; kk += 32) {
                v16bf a  = load_a_rows(Qb, DK, tm, kk, lane);
                v16bf bb = load_b_rows(SbT, DK, tn, kk, lane);
                acc = wmma_bf16(a, bb, acc);
            }
#pragma unroll
            for (int j = 0; j < 8; ++j) acc[j] *= Gc[tm + j + hi * 8];
#pragma unroll
            for (int kk = 0; kk < CHUNK; kk += 32) {
                v16bf a  = load_a_rows(Pb, CHUNK, tm, kk, lane);
                v16bf bb = load_b_rows(UbT, CHUNK, tn, kk, lane);
                acc = wmma_bf16(a, bb, acc);
            }
#pragma unroll
            for (int j = 0; j < 8; ++j) {
                int t = tm + j + hi * 8, vc = tn + n15;
                gout[(((size_t)b * T + (c0 + t)) * H + h) * DV + dv0 + vc] = acc[j];
            }
        }
        __syncthreads();   // all reads of SbT complete before state update

        // ---- phase 8: S = g_last*S + K~^T @ U ----
        {
            float gl = Gc[CHUNK - 1];
#pragma unroll
            for (int ti = 0; ti < 4; ++ti) {
                int tile = wave * 4 + ti;
                int tm = (tile >> 2) * 16;   // d offset (0..112)
                int tn = (tile & 3) * 16;    // v offset (0..48)
                v8f acc = zero8();
#pragma unroll
                for (int kk = 0; kk < CHUNK; kk += 32) {
                    v16bf a  = load_a_rows(Ktld, CHUNK, tm, kk, lane);
                    v16bf bb = load_b_rows(UbT, CHUNK, tn, kk, lane);
                    acc = wmma_bf16(a, bb, acc);
                }
#pragma unroll
                for (int j = 0; j < 8; ++j) {
                    int d = tm + j + hi * 8, vc = tn + n15;
                    float ns = gl * S[d * DVT + vc] + acc[j];
                    S[d * DVT + vc] = ns;
                    SbT[vc * DK + d] = (bf16)ns;
                }
            }
        }
        __syncthreads();
    }

    // ---- final state out (vectorized) ----
    {
        float* sf = gsfinal + ((size_t)(b * H + h)) * DK * DV + dv0;
        for (int i = tid; i < DK * DVT / 4; i += NTHREADS) {
            int d  = i / (DVT / 4);
            int vq = (i % (DVT / 4)) * 4;
            const float* sp = S + d * DVT + vq;
            float4 x;
            x.x = sp[0]; x.y = sp[1]; x.z = sp[2]; x.w = sp[3];
            *(float4*)(sf + (size_t)d * DV + vq) = x;
        }
    }
}

extern "C" void kernel_launch(void* const* d_in, const int* in_sizes, int n_in,
                              void* d_out, int out_size, void* d_ws, size_t ws_size,
                              hipStream_t stream) {
    (void)in_sizes; (void)n_in; (void)out_size; (void)d_ws; (void)ws_size;
    const float* q     = (const float*)d_in[0];
    const float* k     = (const float*)d_in[1];
    const float* v     = (const float*)d_in[2];
    const float* alpha = (const float*)d_in[3];
    const float* beta  = (const float*)d_in[4];
    const float* s0    = (const float*)d_in[5];

    constexpr int B = 2, T = 4096, H = 16;
    float* out    = (float*)d_out;
    float* sfinal = out + (size_t)B * T * H * DV;   // outs then final state, flat

    dim3 grid(B * H * 2);   // Dv split by 2 -> 64 independent workgroups
    dim3 block(NTHREADS);
    hipLaunchKernelGGL(gdn_chunked_wmma_kernel, grid, block, 0, stream,
                       q, k, v, alpha, beta, s0, out, sfinal);
}
